// _System_robust_passive_output_54305566490949
// MI455X (gfx1250) — compile-verified, matrix operand order checked by reading, not run on hardware
//
#include <hip/hip_runtime.h>
#include <math.h>

// Sizes from the reference
#define NXc 256
#define NYc 128
#define NUc 128
#define NQc 512
#define Bc  32768
#define EPSV 0.01f
// RHO = 0.5  -> 0.5/RHO = 1.0, 2*RHO = 1.0 ; ALPHA = 0.0  (constants folded below)

typedef __attribute__((ext_vector_type(16))) __bf16 v16bf;
typedef __attribute__((ext_vector_type(8)))  float  v8f;

__device__ inline unsigned short f2bf(float x) {
  unsigned int u = __float_as_uint(x);
  return (unsigned short)((u + 0x7fffu + ((u >> 16) & 1u)) >> 16);
}
__device__ inline float bf2f(unsigned short h) {
  return __uint_as_float(((unsigned int)h) << 16);
}

// Load a 16x32 bf16 fragment (A-pattern; also valid as B for C = A @ W^T with W[N,K] row-major).
// lanes 0-15: row = row0+lane, K = {k0..k0+7, k0+16..k0+23}
// lanes 16-31: row = row0+lane-16, K = {k0+8..k0+15, k0+24..k0+31}
// Requires ld % 8 == 0 and k0 % 32 == 0 (all call sites satisfy this -> 16B aligned loads).
__device__ inline v16bf load_frag(const unsigned short* __restrict__ base, int ld, int row0, int k0) {
  int lane = threadIdx.x & 31;
  int r  = row0 + (lane & 15);
  int ks = k0 + ((lane >> 4) << 3);
  const unsigned short* p = base + (size_t)r * ld + ks;
  uint4 lo = *reinterpret_cast<const uint4*>(p);
  uint4 hi = *reinterpret_cast<const uint4*>(p + 16);
  v16bf f;
  __builtin_memcpy(&f, &lo, 16);
  __builtin_memcpy(reinterpret_cast<char*>(&f) + 16, &hi, 16);
  return f;
}

__device__ inline v8f wmma_bf16(v16bf a, v16bf b, v8f c) {
  return __builtin_amdgcn_wmma_f32_16x16x32_bf16(false, a, false, b, (short)0, c, false, false);
}

// ---------------- Phase A: small parameter math (naive fp32, perf-irrelevant) ---------------

__global__ void k_P(const float* __restrict__ Pstar, float* __restrict__ P) {
  int idx = blockIdx.x * blockDim.x + threadIdx.x;
  if (idx >= NXc * NXc) return;
  int i = idx / NXc, j = idx % NXc;
  float s = 0.f;
  for (int k = 0; k < NXc; ++k) s += Pstar[i * NXc + k] * Pstar[j * NXc + k];
  P[idx] = 0.5f * s + (i == j ? EPSV : 0.f);
}

__global__ void k_M(const float* __restrict__ X3, const float* __restrict__ Y3, float* __restrict__ M) {
  int idx = blockIdx.x * blockDim.x + threadIdx.x;
  if (idx >= NUc * NUc) return;
  int i = idx / NUc, j = idx % NUc;
  float s = 0.f;
  for (int k = 0; k < NUc; ++k) s += X3[i * NUc + k] * X3[j * NUc + k];
  M[idx] = s + Y3[i * NUc + j] - Y3[j * NUc + i] + (i == j ? EPSV : 0.f);
}

// aug = [A (+I if addI) | I], n x 2n
__global__ void k_aug(const float* __restrict__ A, float* __restrict__ aug, int n, int addI) {
  int idx = blockIdx.x * blockDim.x + threadIdx.x;
  int w = 2 * n;
  if (idx >= n * w) return;
  int i = idx / w, j = idx % w;
  float v;
  if (j < n) v = A[i * n + j] + ((addI && i == j) ? 1.f : 0.f);
  else       v = ((j - n) == i) ? 1.f : 0.f;
  aug[idx] = v;
}

// Single-workgroup Gauss-Jordan (no pivoting; matrices have PD symmetric parts)
__global__ void k_gj(float* __restrict__ aug, int n) {
  __shared__ float fac[256];
  __shared__ float piv;
  int tid = threadIdx.x, tn = blockDim.x;
  int w = 2 * n;
  for (int k = 0; k < n; ++k) {
    if (tid == 0) piv = 1.0f / aug[(size_t)k * w + k];
    __syncthreads();
    float pv = piv;
    for (int j = tid; j < w; j += tn) aug[(size_t)k * w + j] *= pv;
    __syncthreads();
    for (int i = tid; i < n; i += tn) fac[i] = (i == k) ? 0.f : aug[(size_t)i * w + k];
    __syncthreads();
    for (int idx = tid; idx < n * w; idx += tn) {
      int i = idx / w, j = idx % w;
      if (i != k) aug[idx] -= fac[i] * aug[(size_t)k * w + j];
    }
    __syncthreads();
  }
}

__global__ void k_extract(const float* __restrict__ aug, float* __restrict__ inv, int n) {
  int idx = blockIdx.x * blockDim.x + threadIdx.x;
  if (idx >= n * n) return;
  int i = idx / n, j = idx % n;
  inv[idx] = aug[(size_t)i * 2 * n + n + j];
}

// D22 = (I - M) @ inv(I+M) + I     (0.5/RHO == 1)
__global__ void k_D22(const float* __restrict__ M, const float* __restrict__ ipm, float* __restrict__ D22) {
  int idx = blockIdx.x * blockDim.x + threadIdx.x;
  if (idx >= NUc * NUc) return;
  int i = idx / NUc, j = idx % NUc;
  float s = (i == j) ? 1.f : 0.f;
  for (int k = 0; k < NUc; ++k) {
    float l = ((i == k) ? 1.f : 0.f) - M[i * NUc + k];
    s += l * ipm[k * NUc + j];
  }
  D22[idx] = s;
}

// R = D22 + D22^T - D22^T @ D22    (2*RHO == 1)
__global__ void k_Rcap(const float* __restrict__ D22, float* __restrict__ R) {
  int idx = blockIdx.x * blockDim.x + threadIdx.x;
  if (idx >= NUc * NUc) return;
  int i = idx / NUc, j = idx % NUc;
  float s = D22[i * NUc + j] + D22[j * NUc + i];
  for (int k = 0; k < NUc; ++k) s -= D22[k * NUc + i] * D22[k * NUc + j];
  R[idx] = s;
}

// vec_VT (768x128) and vec_CD (768x128)
__global__ void k_VTCD(const float* __restrict__ P, const float* __restrict__ B2,
                       const float* __restrict__ C2, const float* __restrict__ D21,
                       const float* __restrict__ DD12, const float* __restrict__ D22,
                       float* __restrict__ VT, float* __restrict__ CD) {
  int idx = blockIdx.x * blockDim.x + threadIdx.x;
  if (idx >= (NXc + NQc) * NUc) return;
  int i = idx / NUc, j = idx % NUc;
  if (i < NXc) {
    float c2t = C2[j * NXc + i];           // C2^T[i,j]
    CD[idx] = c2t;
    float a = 0.f;
    for (int k = 0; k < NXc; ++k) a += P[i * NXc + k] * B2[k * NUc + j];
    float c = 0.f;
    for (int k = 0; k < NYc; ++k) c += C2[k * NXc + i] * D22[k * NUc + j];
    VT[idx] = -a + c2t - c;
  } else {
    int r = i - NXc;
    float d21t = D21[j * NQc + r];         // D21^T[r,j]
    CD[idx] = d21t;
    float c = 0.f;
    for (int k = 0; k < NYc; ++k) c += D21[k * NQc + r] * D22[k * NUc + j];
    VT[idx] = d21t - DD12[r * NUc + j] - c;
  }
}

__global__ void k_W1(const float* __restrict__ VT, const float* __restrict__ Rinv, float* __restrict__ W1) {
  int idx = blockIdx.x * blockDim.x + threadIdx.x;
  if (idx >= (NXc + NQc) * NUc) return;
  int i = idx / NUc, j = idx % NUc;
  float s = 0.f;
  for (int k = 0; k < NUc; ++k) s += VT[i * NUc + k] * Rinv[k * NUc + j];
  W1[idx] = s;
}

// H = X X^T + eps I + W1 VT^T + CD CD^T   (2*RHO == 1)
__global__ void k_H(const float* __restrict__ X, const float* __restrict__ W1,
                    const float* __restrict__ VT, const float* __restrict__ CD,
                    float* __restrict__ H) {
  const int n = NXc + NQc;
  int idx = blockIdx.x * blockDim.x + threadIdx.x;
  if (idx >= n * n) return;
  int i = idx / n, j = idx % n;
  float s = (i == j) ? EPSV : 0.f;
  for (int k = 0; k < n; ++k) s += X[i * n + k] * X[j * n + k];
  for (int k = 0; k < NUc; ++k) s += W1[i * NUc + k] * VT[j * NUc + k];
  for (int k = 0; k < NUc; ++k) s += CD[i * NUc + k] * CD[j * NUc + k];
  H[idx] = s;
}

// Y = -0.5*(H1 + Y1 - Y1^T)  (ALPHA==0);  lam[q] = 0.5*H4[q,q]
__global__ void k_Ylam(const float* __restrict__ H, const float* __restrict__ Y1,
                       float* __restrict__ Y, float* __restrict__ lam) {
  const int n = NXc + NQc;
  int idx = blockIdx.x * blockDim.x + threadIdx.x;
  if (idx < NXc * NXc) {
    int i = idx / NXc, j = idx % NXc;
    Y[idx] = -0.5f * (H[i * n + j] + Y1[i * NXc + j] - Y1[j * NXc + i]);
  }
  if (idx < NQc) lam[idx] = 0.5f * H[(size_t)(NXc + idx) * n + NXc + idx];
}

__global__ void k_A(const float* __restrict__ Pinv, const float* __restrict__ Y, float* __restrict__ A_) {
  int idx = blockIdx.x * blockDim.x + threadIdx.x;
  if (idx >= NXc * NXc) return;
  int i = idx / NXc, j = idx % NXc;
  float s = 0.f;
  for (int k = 0; k < NXc; ++k) s += Pinv[i * NXc + k] * Y[k * NXc + j];
  A_[idx] = s;
}

// B1 = Pinv @ (-H2 - Chi)
__global__ void k_B1(const float* __restrict__ Pinv, const float* __restrict__ H,
                     const float* __restrict__ Chi, float* __restrict__ B1) {
  const int n = NXc + NQc;
  int idx = blockIdx.x * blockDim.x + threadIdx.x;
  if (idx >= NXc * NQc) return;
  int i = idx / NQc, q = idx % NQc;
  float s = 0.f;
  for (int k = 0; k < NXc; ++k)
    s += Pinv[i * NXc + k] * (-H[(size_t)k * n + NXc + q] - Chi[k * NQc + q]);
  B1[idx] = s;
}

// Wbase[q, 0:384] = [C1[q,:]=Chi^T/lam | D12[q,:]=DD12/lam]   (bf16)
__global__ void k_packWbase(const float* __restrict__ Chi, const float* __restrict__ DD12,
                            const float* __restrict__ lam, unsigned short* __restrict__ Wb) {
  int idx = blockIdx.x * blockDim.x + threadIdx.x;
  if (idx >= NQc * (NXc + NUc)) return;
  int q = idx / (NXc + NUc), k = idx % (NXc + NUc);
  float v = (k < NXc) ? Chi[k * NQc + q] : DD12[q * NUc + (k - NXc)];
  Wb[idx] = f2bf(v / lam[q]);
}

// Wout[i, 0:896] = [A[i,:256] | B2[i,:128] | B1[i,:512]]   (bf16)
__global__ void k_packWout(const float* __restrict__ A_, const float* __restrict__ B2,
                           const float* __restrict__ B1, unsigned short* __restrict__ Wo) {
  const int K = NXc + NUc + NQc; // 896
  int idx = blockIdx.x * blockDim.x + threadIdx.x;
  if (idx >= NXc * K) return;
  int i = idx / K, k = idx % K;
  float v = (k < NXc) ? A_[i * NXc + k]
          : (k < NXc + NUc) ? B2[i * NUc + (k - NXc)]
          : B1[i * NQc + (k - NXc - NUc)];
  Wo[idx] = f2bf(v);
}

// D11 = -tril(H4,-1)/lam[:,None]  -> fp32 (for serial in-block) + bf16 (for WMMA updates)
__global__ void k_packD11(const float* __restrict__ H, const float* __restrict__ lam,
                          float* __restrict__ D11f, unsigned short* __restrict__ D11bf) {
  const int n = NXc + NQc;
  int idx = blockIdx.x * blockDim.x + threadIdx.x;
  if (idx >= NQc * NQc) return;
  int i = idx / NQc, j = idx % NQc;
  float v = (i > j) ? (-H[(size_t)(NXc + i) * n + NXc + j] / lam[i]) : 0.f;
  D11f[idx] = v;
  D11bf[idx] = f2bf(v);
}

// XU[r, 0:384] = [xi[r,:] | u[r,:]]   (bf16)
__global__ void k_packXU(const float* __restrict__ xi, const float* __restrict__ u,
                         unsigned short* __restrict__ XU) {
  int idx = blockIdx.x * blockDim.x + threadIdx.x;
  if (idx >= Bc * (NXc + NUc)) return;
  int r = idx / (NXc + NUc), k = idx % (NXc + NUc);
  float v = (k < NXc) ? xi[(size_t)r * NXc + k] : u[(size_t)r * NUc + (k - NXc)];
  XU[idx] = f2bf(v);
}

__global__ void k_zero16(unsigned short* __restrict__ p, int n) {
  int idx = blockIdx.x * blockDim.x + threadIdx.x;
  if (idx < n) p[idx] = 0;
}

// ---------------- Phase B: batch WMMA kernels ---------------------------------------------

// out = [A0 | A1] @ W^T, bf16 in / f32 WMMA accumulate. W is [N, K0+K1] row-major.
// Register-blocked: each wave owns one 16-row M tile and NB=4 consecutive 16-col N tiles,
// reusing each A fragment across 4 WMMAs (2.5 b128 loads per wmma instead of 4).
#define NB 4
__global__ void k_wmma_gemm(const unsigned short* __restrict__ A0, int lda0, int K0,
                            const unsigned short* __restrict__ A1, int lda1, int K1,
                            const unsigned short* __restrict__ W, int ldw,
                            float* __restrict__ outF, int ldc, int ngroups) {
  int wave = threadIdx.x >> 5;
  int lane = threadIdx.x & 31;
  int wid = blockIdx.x * (blockDim.x >> 5) + wave;
  int mt = wid / ngroups, ng = wid % ngroups;
  int m0 = mt * 16, n0 = ng * (NB * 16);
  v8f acc[NB];
#pragma unroll
  for (int j = 0; j < NB; ++j) acc[j] = (v8f){};
  for (int k0 = 0; k0 < K0; k0 += 32) {
    v16bf a = load_frag(A0, lda0, m0, k0);
#pragma unroll
    for (int j = 0; j < NB; ++j) {
      v16bf b = load_frag(W, ldw, n0 + j * 16, k0);
      acc[j] = wmma_bf16(a, b, acc[j]);
    }
  }
  for (int k0 = 0; k0 < K1; k0 += 32) {
    v16bf a = load_frag(A1, lda1, m0, k0);
#pragma unroll
    for (int j = 0; j < NB; ++j) {
      v16bf b = load_frag(W, ldw, n0 + j * 16, K0 + k0);
      acc[j] = wmma_bf16(a, b, acc[j]);
    }
  }
  int rb = m0 + ((lane >> 4) << 3);
#pragma unroll
  for (int j = 0; j < NB; ++j) {
    int cc = n0 + j * 16 + (lane & 15);
#pragma unroll
    for (int v = 0; v < 8; ++v) outF[(size_t)(rb + v) * ldc + cc] = acc[j][v];
  }
}

// Blocked implicit triangular solve with tanh.
//   Each wave owns 32 batch rows (two 16-row C tiles sharing every B fragment),
//   marches over the 32 column blocks of 16:
//     acc = base tile (fused GEMM over XU/Wbase, computed exactly once per (tile,block))
//         + WMMA updates over previously produced bf16 w (zero-padded -> odd-16 K-tail exact),
//   then a 16-wide in-block serial tanh recursion in fp32 with ALL 32 lanes busy
//   (lane r handles batch row m0+r).
__global__ void k_solve(const unsigned short* __restrict__ XU, const unsigned short* __restrict__ Wbase,
                        const unsigned short* __restrict__ D11bf, const float* __restrict__ D11f,
                        unsigned short* __restrict__ wbf) {
  __shared__ float dsm[16][17];      // shared diagonal D11 block (fp32)
  __shared__ float sm[2][32][17];    // per-wave acc tiles (32 rows)
  __shared__ float wsm[2][32][17];   // per-wave in-block w (fp32)
  int wave = threadIdx.x >> 5;       // 2 waves per block
  int lane = threadIdx.x & 31;
  int m0 = (blockIdx.x * 4 + wave * 2) * 16;   // 32 rows per wave, 64 per block

  for (int b = 0; b < 32; ++b) {
    int b16 = b * 16;
    // cooperative load of the diagonal block (uniform across waves): 64 threads x 4 elems
#pragma unroll
    for (int t = 0; t < 4; ++t) {
      int idx = threadIdx.x + t * 64;
      dsm[idx >> 4][idx & 15] = D11f[(size_t)(b16 + (idx >> 4)) * NQc + b16 + (idx & 15)];
    }
    __syncthreads();

    v8f acc0 = {}, acc1 = {};
    // base tiles: [xi|u] @ Wbase^T   (K = 384) ; B fragment shared by both M tiles
    for (int k0 = 0; k0 < NXc + NUc; k0 += 32) {
      v16bf bb = load_frag(Wbase, NXc + NUc, b16, k0);
      v16bf a0 = load_frag(XU, NXc + NUc, m0, k0);
      v16bf a1 = load_frag(XU, NXc + NUc, m0 + 16, k0);
      acc0 = wmma_bf16(a0, bb, acc0);
      acc1 = wmma_bf16(a1, bb, acc1);
    }
    // cross-block update: w_prev @ D11_block^T
    for (int k0 = 0; k0 < b16; k0 += 32) {
      v16bf bb = load_frag(D11bf, NQc, b16, k0);
      v16bf a0 = load_frag(wbf, NQc, m0, k0);
      v16bf a1 = load_frag(wbf, NQc, m0 + 16, k0);
      acc0 = wmma_bf16(a0, bb, acc0);
      acc1 = wmma_bf16(a1, bb, acc1);
    }

    int rb = (lane >> 4) << 3, cc = lane & 15;
#pragma unroll
    for (int v = 0; v < 8; ++v) {
      sm[wave][rb + v][cc]      = acc0[v];
      sm[wave][16 + rb + v][cc] = acc1[v];
    }
    __syncthreads();  // make tiles visible across lanes of each wave

    { // serial in-block recursion: one lane per batch row (all 32 lanes active)
      int r = lane;
      for (int j = 0; j < 16; ++j) {
        float vv = sm[wave][r][j];
        for (int t = 0; t < j; ++t) vv += dsm[j][t] * wsm[wave][r][t];
        float wv = tanhf(vv);
        wsm[wave][r][j] = wv;
        wbf[(size_t)(m0 + r) * NQc + b16 + j] = f2bf(wv);
      }
    }
    __syncthreads();  // protect dsm before next block overwrites it
  }
}

// ---------------- Host launcher ------------------------------------------------------------

extern "C" void kernel_launch(void* const* d_in, const int* in_sizes, int n_in,
                              void* d_out, int out_size, void* d_ws, size_t ws_size,
                              hipStream_t stream) {
  const float* xi    = (const float*)d_in[0];
  const float* u     = (const float*)d_in[1];
  const float* Pstar = (const float*)d_in[2];
  const float* Chi   = (const float*)d_in[3];
  const float* DD12  = (const float*)d_in[4];
  const float* X3    = (const float*)d_in[5];
  const float* Y3    = (const float*)d_in[6];
  const float* Y1    = (const float*)d_in[7];
  const float* C2    = (const float*)d_in[8];
  const float* D21   = (const float*)d_in[9];
  const float* B2    = (const float*)d_in[10];
  const float* X     = (const float*)d_in[11];
  float* out = (float*)d_out;

  char* ws = (char*)d_ws;
  size_t off = 0;
  auto alloc = [&](size_t bytes) -> char* {
    char* p = ws + off;
    off += (bytes + 255) & ~(size_t)255;
    return p;
  };

  float* P     = (float*)alloc((size_t)NXc * NXc * 4);
  float* M     = (float*)alloc((size_t)NUc * NUc * 4);
  float* aug   = (float*)alloc((size_t)NXc * 2 * NXc * 4);   // reused for all 3 inversions
  float* ipm   = (float*)alloc((size_t)NUc * NUc * 4);
  float* D22   = (float*)alloc((size_t)NUc * NUc * 4);
  float* Rcap  = (float*)alloc((size_t)NUc * NUc * 4);
  float* Rinv  = (float*)alloc((size_t)NUc * NUc * 4);
  float* VT    = (float*)alloc((size_t)(NXc + NQc) * NUc * 4);
  float* CD    = (float*)alloc((size_t)(NXc + NQc) * NUc * 4);
  float* W1    = (float*)alloc((size_t)(NXc + NQc) * NUc * 4);
  float* H     = (float*)alloc((size_t)(NXc + NQc) * (NXc + NQc) * 4);
  float* Pinv  = (float*)alloc((size_t)NXc * NXc * 4);
  float* Yb    = (float*)alloc((size_t)NXc * NXc * 4);
  float* A_    = (float*)alloc((size_t)NXc * NXc * 4);
  float* B1    = (float*)alloc((size_t)NXc * NQc * 4);
  float* lam   = (float*)alloc((size_t)NQc * 4);
  float* D11f  = (float*)alloc((size_t)NQc * NQc * 4);
  unsigned short* Wbase = (unsigned short*)alloc((size_t)NQc * (NXc + NUc) * 2);
  unsigned short* Wout  = (unsigned short*)alloc((size_t)NXc * (NXc + NUc + NQc) * 2);
  unsigned short* D11bf = (unsigned short*)alloc((size_t)NQc * NQc * 2);
  unsigned short* XU    = (unsigned short*)alloc((size_t)Bc * (NXc + NUc) * 2);
  unsigned short* wbf   = (unsigned short*)alloc((size_t)Bc * NQc * 2);
  (void)ws_size; (void)in_sizes; (void)n_in; (void)out_size;

  const int T = 256;
  auto g = [](int n) { return (n + 255) / 256; };

  // ---- Phase A: parameter update ----
  k_P<<<g(NXc * NXc), T, 0, stream>>>(Pstar, P);
  k_M<<<g(NUc * NUc), T, 0, stream>>>(X3, Y3, M);

  // inv(I + M)
  k_aug<<<g(NUc * 2 * NUc), T, 0, stream>>>(M, aug, NUc, 1);
  k_gj<<<1, T, 0, stream>>>(aug, NUc);
  k_extract<<<g(NUc * NUc), T, 0, stream>>>(aug, ipm, NUc);

  k_D22<<<g(NUc * NUc), T, 0, stream>>>(M, ipm, D22);
  k_Rcap<<<g(NUc * NUc), T, 0, stream>>>(D22, Rcap);

  // inv(R_cap)
  k_aug<<<g(NUc * 2 * NUc), T, 0, stream>>>(Rcap, aug, NUc, 0);
  k_gj<<<1, T, 0, stream>>>(aug, NUc);
  k_extract<<<g(NUc * NUc), T, 0, stream>>>(aug, Rinv, NUc);

  k_VTCD<<<g((NXc + NQc) * NUc), T, 0, stream>>>(P, B2, C2, D21, DD12, D22, VT, CD);
  k_W1<<<g((NXc + NQc) * NUc), T, 0, stream>>>(VT, Rinv, W1);
  k_H<<<g((NXc + NQc) * (NXc + NQc)), T, 0, stream>>>(X, W1, VT, CD, H);

  // inv(P)
  k_aug<<<g(NXc * 2 * NXc), T, 0, stream>>>(P, aug, NXc, 0);
  k_gj<<<1, T, 0, stream>>>(aug, NXc);
  k_extract<<<g(NXc * NXc), T, 0, stream>>>(aug, Pinv, NXc);

  k_Ylam<<<g(NXc * NXc), T, 0, stream>>>(H, Y1, Yb, lam);
  k_A<<<g(NXc * NXc), T, 0, stream>>>(Pinv, Yb, A_);
  k_B1<<<g(NXc * NQc), T, 0, stream>>>(Pinv, H, Chi, B1);

  // pack bf16 operands
  k_packWbase<<<g(NQc * (NXc + NUc)), T, 0, stream>>>(Chi, DD12, lam, Wbase);
  k_packWout<<<g(NXc * (NXc + NUc + NQc)), T, 0, stream>>>(A_, B2, B1, Wout);
  k_packD11<<<g(NQc * NQc), T, 0, stream>>>(H, lam, D11f, D11bf);
  k_packXU<<<g(Bc * (NXc + NUc)), T, 0, stream>>>(xi, u, XU);
  k_zero16<<<g(Bc * NQc), T, 0, stream>>>(wbf, Bc * NQc);

  // ---- Phase B: batch (WMMA) ----
  // w-solve (base GEMM fused): 2048 row tiles, 2 waves x 2 tiles per block -> 512 blocks
  k_solve<<<Bc / 64, 64, 0, stream>>>(XU, Wbase, D11bf, D11f, wbf);

  // out = [xi|u|w] @ [A|B2|B1]^T   (f32 output), NB=4 column tiles per wave
  {
    int ngroups = NXc / 16 / NB;                 // 4 groups of 4 column tiles
    int wavesTotal = (Bc / 16) * ngroups;        // 8192 waves
    k_wmma_gemm<<<wavesTotal / 8, T, 0, stream>>>(XU, NXc + NUc, NXc + NUc,
                                                  wbf, NQc, NQc,
                                                  Wout, NXc + NUc + NQc,
                                                  out, NXc, ngroups);
  }
}
